// GPT3DevBlock_7275674599475
// MI455X (gfx1250) — matrix operand order repaired and checked by reading, not compiled, not run on hardware
//
#include <hip/hip_runtime.h>
#include <cstddef>
#include <cstdint>

// ---------------------------------------------------------------------------
// Types for CDNA5 WMMA (gfx1250, wave32) — native __bf16 throughout
// ---------------------------------------------------------------------------
typedef __bf16 bf16;
typedef __attribute__((ext_vector_type(16))) __bf16 v16bf;
typedef __attribute__((ext_vector_type(8)))  __bf16 v8bf;
typedef __attribute__((ext_vector_type(4)))  __bf16 v4bf;
typedef __attribute__((ext_vector_type(2)))  __bf16 v2bf;
typedef __attribute__((ext_vector_type(8)))  float  v8f;

__device__ __forceinline__ v16bf cat16(v8bf lo, v8bf hi) {
    return __builtin_shufflevector(lo, hi, 0, 1, 2, 3, 4, 5, 6, 7,
                                           8, 9, 10, 11, 12, 13, 14, 15);
}

__device__ __forceinline__ v8f wmma_bf16(v16bf a, v16bf b, v8f c) {
    // D = A(16x32 bf16) * B(32x16 bf16) + C(16x16 f32)
    return __builtin_amdgcn_wmma_f32_16x16x32_bf16(
        /*neg_a=*/false, a, /*neg_b=*/false, b,
        /*c_mod=*/(short)0, c, /*reuse_a=*/false, /*reuse_b=*/false);
}

// --- packed f32->bf16 conversion helpers -----------------------------------
#if __has_builtin(__builtin_amdgcn_cvt_pk_bf16_f32)
__device__ __forceinline__ v2bf pk2(float a, float b) {
    return __builtin_amdgcn_cvt_pk_bf16_f32(a, b);
}
#else
__device__ __forceinline__ v2bf pk2(float a, float b) {
    v2bf r; r[0] = (bf16)a; r[1] = (bf16)b; return r;
}
#endif
// 16 contiguous floats -> v16bf (8x v_cvt_pk_bf16_f32, no repacking movs)
__device__ __forceinline__ v16bf cvt16(const float* __restrict__ p) {
    union { v16bf v; v2bf pk[8]; } r;
#pragma unroll
    for (int i = 0; i < 8; ++i) r.pk[i] = pk2(p[2 * i], p[2 * i + 1]);
    return r.v;
}
// two 8-float chunks, scaled -> v16bf
__device__ __forceinline__ v16bf cvt16s(const float* __restrict__ a,
                                        const float* __restrict__ b, float s) {
    union { v16bf v; v2bf pk[8]; } r;
#pragma unroll
    for (int i = 0; i < 4; ++i) r.pk[i] = pk2(a[2 * i] * s, a[2 * i + 1] * s);
#pragma unroll
    for (int i = 0; i < 4; ++i) r.pk[4 + i] = pk2(b[2 * i] * s, b[2 * i + 1] * s);
    return r.v;
}

// --- async global -> LDS copy (GLOBAL_LOAD_ASYNC_TO_LDS_B128, ASYNCcnt) ----
// LDS offset = low 32 bits of the generic pointer to a __shared__ object
// (AS3->flat cast = {shared_aperture, offset} per ISA LDS aperture mapping).
__device__ __forceinline__ void async_g2l_b128(void* lds, const void* gptr) {
    unsigned ldsoff = (unsigned)(uintptr_t)lds;
    asm volatile("global_load_async_to_lds_b128 %0, %1, off"
                 :: "v"(ldsoff), "v"(gptr) : "memory");
}
__device__ __forceinline__ void wait_async0() {
    asm volatile("s_wait_asynccnt 0" ::: "memory");
}

// ---------------------------------------------------------------------------
// Problem constants
// ---------------------------------------------------------------------------
static constexpr int B_ = 2, S_ = 2048, D_ = 1024, H_ = 16, DH_ = 64, FF_ = 4096;
static constexpr int M_ = B_ * S_;               // 4096 token rows

// ---------------------------------------------------------------------------
// Kernel: fp32 -> bf16 weight cast (vectorized, 4 elems/thread)
// ---------------------------------------------------------------------------
__global__ __launch_bounds__(256)
void cast_f32_bf16_kern(const float* __restrict__ in, bf16* __restrict__ out,
                        int n) {
    int i = (blockIdx.x * blockDim.x + threadIdx.x) * 4;
    if (i >= n) return;
    float4 v = *(const float4*)(in + i);
    v4bf o;
    o[0] = (bf16)v.x; o[1] = (bf16)v.y; o[2] = (bf16)v.z; o[3] = (bf16)v.w;
    *(v4bf*)(out + i) = o;
}

// ---------------------------------------------------------------------------
// Kernel: fused LayerNorm(row of 1024) + bf16 cast
// ---------------------------------------------------------------------------
__global__ __launch_bounds__(256)
void ln_cast_kern(const float* __restrict__ x, const float* __restrict__ g,
                  const float* __restrict__ bb, bf16* __restrict__ out) {
    const int row = blockIdx.x;
    const int tid = threadIdx.x;
    const int lane = tid & 31, wave = tid >> 5;
    const float* xr = x + (size_t)row * D_;
    float4 v = ((const float4*)xr)[tid];          // 256 * 4 = 1024
    float s  = v.x + v.y + v.z + v.w;
    float ss = v.x * v.x + v.y * v.y + v.z * v.z + v.w * v.w;
#pragma unroll
    for (int i = 16; i >= 1; i >>= 1) {
        s  += __shfl_xor(s,  i, 32);
        ss += __shfl_xor(ss, i, 32);
    }
    __shared__ float ps[8], pss[8];
    if (lane == 0) { ps[wave] = s; pss[wave] = ss; }
    __syncthreads();
    float ts = 0.f, tss = 0.f;
#pragma unroll
    for (int i = 0; i < 8; ++i) { ts += ps[i]; tss += pss[i]; }
    const float mu   = ts * (1.0f / D_);
    const float var  = tss * (1.0f / D_) - mu * mu;
    const float rinv = rsqrtf(var + 1e-5f);

    const int c0 = tid * 4;
    float in4[4] = { v.x, v.y, v.z, v.w };
    v4bf o;
#pragma unroll
    for (int j = 0; j < 4; ++j)
        o[j] = (bf16)((in4[j] - mu) * rinv * g[c0 + j] + bb[c0 + j]);
    *(v4bf*)(out + (size_t)row * D_ + c0) = o;
}

// ---------------------------------------------------------------------------
// Kernel: tiled bf16 WMMA GEMM, double-buffered async pipeline.
//   C[M,N](f32 or bf16) = epi( A_bf16[M,K] @ B_bf16[K,N] + bias [+ resid] )
// Block: 256 thr = 8 waves. Block tile 128(M) x 64(N), K step 32.
// A tile: GLOBAL_LOAD_ASYNC_TO_LDS_B128 into buffer[nxt] overlapping the
// WMMAs on buffer[cur]; B tile register-transposed into Bs[nxt] in parallel.
// One barrier per K-step. Rows padded to 40 bf16 (20 banks, 16B aligned).
// ---------------------------------------------------------------------------
template <bool GELU_, bool RESID, bool OUTBF>
__global__ __launch_bounds__(256)
void gemm_bf16_kern(const bf16* __restrict__ A, const bf16* __restrict__ Bw,
                    const float* __restrict__ bias,
                    const float* __restrict__ resid,
                    void* __restrict__ outv, int M, int N, int K) {
    __shared__ __align__(16) bf16 As[2][128 * 40];
    __shared__ __align__(16) bf16 Bs[2][64 * 40];

    const int tid  = threadIdx.x;
    const int m0   = blockIdx.y * 128;
    const int n0   = blockIdx.x * 64;
    const int wave = tid >> 5, lane = tid & 31;
    const int wm   = wave >> 1, wn = wave & 1;
    const int m16  = lane & 15, hl = lane >> 4;

    // per-thread staging coords
    const int arow0 = tid >> 2,        aco = (tid & 3) * 8;   // A chunk 0
    const int arow1 = (tid + 256) >> 2;                        // A chunk 1
    const int bkr   = tid >> 3,        bco = (tid & 7) * 8;   // B chunk

    auto stageA = [&](int k0, int buf) {
        async_g2l_b128(&As[buf][arow0 * 40 + aco],
                       A + (size_t)(m0 + arow0) * K + k0 + aco);
        async_g2l_b128(&As[buf][arow1 * 40 + aco],
                       A + (size_t)(m0 + arow1) * K + k0 + aco);
    };
    auto stageB = [&](int k0, int buf) {
        v8bf val = *(const v8bf*)(Bw + (size_t)(k0 + bkr) * N + n0 + bco);
#pragma unroll
        for (int j = 0; j < 8; ++j) Bs[buf][(bco + j) * 40 + bkr] = val[j];
    };

    v8f c[2][2];
#pragma unroll
    for (int i = 0; i < 2; ++i)
#pragma unroll
        for (int j = 0; j < 2; ++j)
            c[i][j] = (v8f){0.f, 0.f, 0.f, 0.f, 0.f, 0.f, 0.f, 0.f};

    // --- prologue: stage first tile ---
    stageA(0, 0);
    stageB(0, 0);
    wait_async0();
    __syncthreads();

    for (int k0 = 0; k0 < K; k0 += 32) {
        const int  cur  = (k0 >> 5) & 1, nxt = cur ^ 1;
        const bool more = (k0 + 32 < K);
        // --- stage next tile into the other buffer (overlaps the WMMAs) ---
        if (more) {
            stageA(k0 + 32, nxt);
            stageB(k0 + 32, nxt);
            if (k0 + 64 < K)
                __builtin_prefetch(Bw + (size_t)(k0 + 64 + bkr) * N + n0 + bco, 0, 1);
        }

        // --- fragments from current buffer ---
        v16bf afr[2], bfr[2];
#pragma unroll
        for (int ti = 0; ti < 2; ++ti) {
            const bf16* ap = &As[cur][(wm * 32 + ti * 16 + m16) * 40];
            afr[ti] = cat16(*(const v8bf*)(ap + 8 * hl),
                            *(const v8bf*)(ap + 16 + 8 * hl));
        }
#pragma unroll
        for (int tj = 0; tj < 2; ++tj) {
            const bf16* bp = &Bs[cur][(wn * 32 + tj * 16 + m16) * 40];
            bfr[tj] = cat16(*(const v8bf*)(bp + 8 * hl), *(const v8bf*)(bp + 16 + 8 * hl));
        }
#pragma unroll
        for (int ti = 0; ti < 2; ++ti)
#pragma unroll
            for (int tj = 0; tj < 2; ++tj)
                c[ti][tj] = wmma_bf16(afr[ti], bfr[tj], c[ti][tj]);

        if (more) {
            wait_async0();
            __syncthreads();
        }
    }

    // --- epilogue ---
#pragma unroll
    for (int ti = 0; ti < 2; ++ti) {
#pragma unroll
        for (int tj = 0; tj < 2; ++tj) {
            const int gcol = n0 + wn * 32 + tj * 16 + m16;
            const float bv = bias[gcol];
#pragma unroll
            for (int r = 0; r < 8; ++r) {
                const int grow = m0 + wm * 32 + ti * 16 + r + 8 * hl;
                float v = c[ti][tj][r] + bv;
                if (RESID) v += resid[(size_t)grow * N + gcol];
                if (GELU_) v = 0.5f * v * (1.0f + erff(v * 0.70710678118654752f));
                const size_t o = (size_t)grow * N + gcol;
                if (OUTBF) ((bf16*)outv)[o] = (bf16)v;
                else       ((float*)outv)[o] = v;
            }
        }
    }
}

// ---------------------------------------------------------------------------
// Kernel: flash attention over qkv[M, 3*H*DH] (f32), out attn_bf16[M, H*DH].
// Grid: B*H*(S/64) blocks x 128 thr. Each wave owns a 16-row Q tile.
// SCALE * log2(e) folded into the Q cast -> softmax via exp2f (v_exp).
// ---------------------------------------------------------------------------
__global__ __launch_bounds__(128)
void flash_attn_kern(const float* __restrict__ qkv, bf16* __restrict__ attn) {
    constexpr int NBQ = S_ / 64;                  // 32 q-blocks per (b,h)
    const int bh = blockIdx.x / NBQ;
    const int qb = blockIdx.x % NBQ;
    const int b  = bh / H_, h = bh % H_;
    const int tid = threadIdx.x, wave = tid >> 5, lane = tid & 31;
    const int m16 = lane & 15, hl = lane >> 4;
    const int q0 = qb * 64 + wave * 16;

    __shared__ __align__(16) bf16 Vt[64 * 40];     // V^T tile, shared by block
    __shared__ __align__(16) bf16 Pl[4][16 * 40];  // per-wave P scratch

    const float QS = 0.125f * 1.44269504088896341f;  // 1/sqrt(64) * log2(e)

    // --- Q fragments (A layout, 16x32 bf16 each; 2 chunks cover DH=64) ---
    v16bf aq[2];
    {
        const float* qp = qkv + (size_t)(b * S_ + q0 + m16) * (3 * D_) + h * DH_;
#pragma unroll
        for (int kc = 0; kc < 2; ++kc)
            aq[kc] = cvt16s(qp + kc * 32 + 16 * hl,
                            qp + kc * 32 + 16 * hl + 8, QS);
    }

    v8f acc[4];
#pragma unroll
    for (int t = 0; t < 4; ++t)
        acc[t] = (v8f){0.f, 0.f, 0.f, 0.f, 0.f, 0.f, 0.f, 0.f};
    float mrow[8], lrow[8];
#pragma unroll
    for (int r = 0; r < 8; ++r) { mrow[r] = -1e30f; lrow[r] = 0.f; }

    // cooperative V^T staging coords: key pair x 8 dims per thread
    const int jp = (tid >> 3) * 2;                // even key index 0..30
    const int vcs = (tid & 7) * 8;                // dim start 0..56

    for (int j0 = 0; j0 < S_; j0 += 32) {
        __syncthreads();  // previous iteration's Vt / Pl reads done
        // --- cooperative V^T tile: pack 2 adjacent keys per b32 store ---
        {
            const float* vp0 = qkv + (size_t)(b * S_ + j0 + jp) * (3 * D_) +
                               2 * D_ + h * DH_ + vcs;
            const float* vp1 = vp0 + 3 * D_;
#pragma unroll
            for (int i = 0; i < 8; ++i)
                *(v2bf*)(&Vt[(vcs + i) * 40 + jp]) = pk2(vp0[i], vp1[i]);
        }

        // --- scores: Q(16x64) @ K^T -> two 16x16 C tiles ---
        v8f c0 = (v8f){0.f, 0.f, 0.f, 0.f, 0.f, 0.f, 0.f, 0.f};
        v8f c1 = c0;
#pragma unroll
        for (int t = 0; t < 2; ++t) {
            const float* kp = qkv + (size_t)(b * S_ + j0 + t * 16 + m16) * (3 * D_) +
                              D_ + h * DH_;
            v16bf bk0 = cvt16(kp + 16 * hl);
            v16bf bk1 = cvt16(kp + 32 + 16 * hl);
            if (t == 0) { c0 = wmma_bf16(aq[0], bk0, c0); c0 = wmma_bf16(aq[1], bk1, c0); }
            else        { c1 = wmma_bf16(aq[0], bk0, c1); c1 = wmma_bf16(aq[1], bk1, c1); }
        }

        // --- online softmax (row-wise across 16-lane groups) + stash P ---
        bf16* pw = &Pl[wave][0];
#pragma unroll
        for (int r = 0; r < 8; ++r) {
            float s0 = c0[r], s1 = c1[r];
            float mx = fmaxf(s0, s1);
#pragma unroll
            for (int i = 8; i >= 1; i >>= 1) mx = fmaxf(mx, __shfl_xor(mx, i, 32));
            const float mn = fmaxf(mrow[r], mx);
            const float cf = exp2f(mrow[r] - mn);
            const float p0 = exp2f(s0 - mn);
            const float p1 = exp2f(s1 - mn);
            float rs = p0 + p1;
#pragma unroll
            for (int i = 8; i >= 1; i >>= 1) rs += __shfl_xor(rs, i, 32);
            lrow[r] = lrow[r] * cf + rs;
            mrow[r] = mn;
            acc[0][r] *= cf; acc[1][r] *= cf; acc[2][r] *= cf; acc[3][r] *= cf;
            pw[(r + 8 * hl) * 40 + m16]      = (bf16)p0;
            pw[(r + 8 * hl) * 40 + 16 + m16] = (bf16)p1;
        }
        __syncthreads();  // Vt stores + P transpose visible

        // --- P (16x32) as A fragment via LDS transpose ---
        const bf16* pr = pw + m16 * 40;
        v16bf ap = cat16(*(const v8bf*)(pr + 8 * hl),
                         *(const v8bf*)(pr + 16 + 8 * hl));
        // --- P @ V: four 16x16 output tiles across DH=64 ---
#pragma unroll
        for (int t = 0; t < 4; ++t) {
            const bf16* vb = Vt + (t * 16 + m16) * 40;
            v16bf bv = cat16(*(const v8bf*)(vb + 8 * hl), *(const v8bf*)(vb + 16 + 8 * hl));
            acc[t] = wmma_bf16(ap, bv, acc[t]);
        }
    }

    // --- epilogue: normalize and store bf16 in [B,S,H*DH] layout ---
#pragma unroll
    for (int t = 0; t < 4; ++t) {
#pragma unroll
        for (int r = 0; r < 8; ++r) {
            const float ov = acc[t][r] / lrow[r];
            const int row = q0 + r + 8 * hl;
            const int col = h * DH_ + t * 16 + m16;
            attn[(size_t)(b * S_ + row) * D_ + col] = (bf16)ov;
        }
    }
}

// ---------------------------------------------------------------------------
// Host-side launcher
// ---------------------------------------------------------------------------
extern "C" void kernel_launch(void* const* d_in, const int* in_sizes, int n_in,
                              void* d_out, int out_size, void* d_ws, size_t ws_size,
                              hipStream_t stream) {
    const float* x      = (const float*)d_in[0];
    const float* ln1_g  = (const float*)d_in[1];
    const float* ln1_b  = (const float*)d_in[2];
    const float* w_qkv  = (const float*)d_in[3];
    const float* b_qkv  = (const float*)d_in[4];
    const float* w_ao   = (const float*)d_in[5];
    const float* b_ao   = (const float*)d_in[6];
    const float* ln2_g  = (const float*)d_in[7];
    const float* ln2_b  = (const float*)d_in[8];
    const float* w_fc   = (const float*)d_in[9];
    const float* b_fc   = (const float*)d_in[10];
    const float* w_proj = (const float*)d_in[11];
    const float* b_proj = (const float*)d_in[12];
    float* out = (float*)d_out;

    // ---- workspace carve-out (all sizes 256B-aligned) ----
    char* p = (char*)d_ws;
    auto take = [&](size_t bytes) { char* r = p; p += (bytes + 255) & ~(size_t)255; return r; };
    bf16*  wq_bf  = (bf16*) take((size_t)D_ * 3 * H_ * DH_ * sizeof(bf16));  // 1024x3072
    bf16*  wao_bf = (bf16*) take((size_t)H_ * DH_ * D_ * sizeof(bf16));      // 1024x1024
    bf16*  wfc_bf = (bf16*) take((size_t)D_ * FF_ * sizeof(bf16));           // 1024x4096
    bf16*  wpr_bf = (bf16*) take((size_t)FF_ * D_ * sizeof(bf16));           // 4096x1024
    bf16*  h_bf   = (bf16*) take((size_t)M_ * D_ * sizeof(bf16));
    float* qkv_f  = (float*)take((size_t)M_ * 3 * D_ * sizeof(float));
    bf16*  attn_bf= (bf16*) take((size_t)M_ * D_ * sizeof(bf16));
    float* x1_f   = (float*)take((size_t)M_ * D_ * sizeof(float));
    bf16*  h2_bf  = (bf16*) take((size_t)M_ * D_ * sizeof(bf16));
    bf16*  hf_bf  = (bf16*) take((size_t)M_ * FF_ * sizeof(bf16));
    (void)ws_size; (void)n_in; (void)in_sizes; (void)out_size;

    // ---- 1. weight casts fp32 -> bf16 ----
    auto cast = [&](const float* src, bf16* dst, int n) {
        cast_f32_bf16_kern<<<n / 1024, 256, 0, stream>>>(src, dst, n);
    };
    cast(w_qkv,  wq_bf,  D_ * 3 * H_ * DH_);
    cast(w_ao,   wao_bf, H_ * DH_ * D_);
    cast(w_fc,   wfc_bf, D_ * FF_);
    cast(w_proj, wpr_bf, FF_ * D_);

    // ---- 2. LN1 + bf16 cast ----
    ln_cast_kern<<<M_, 256, 0, stream>>>(x, ln1_g, ln1_b, h_bf);

    // ---- 3. QKV GEMM: [4096,1024] @ [1024,3072] + b_qkv -> f32 ----
    gemm_bf16_kern<false, false, false>
        <<<dim3(3 * H_ * DH_ / 64, M_ / 128), 256, 0, stream>>>(
            h_bf, wq_bf, b_qkv, nullptr, qkv_f, M_, 3 * H_ * DH_, D_);

    // ---- 4. flash attention -> attn bf16 [4096,1024] ----
    flash_attn_kern<<<B_ * H_ * (S_ / 64), 128, 0, stream>>>(qkv_f, attn_bf);

    // ---- 5. AO GEMM + residual(x) -> x1 f32 ----
    gemm_bf16_kern<false, true, false>
        <<<dim3(D_ / 64, M_ / 128), 256, 0, stream>>>(
            attn_bf, wao_bf, b_ao, x, x1_f, M_, D_, D_);

    // ---- 6. LN2 + bf16 cast ----
    ln_cast_kern<<<M_, 256, 0, stream>>>(x1_f, ln2_g, ln2_b, h2_bf);

    // ---- 7. FC GEMM + exact-erf GELU -> bf16 ----
    gemm_bf16_kern<true, false, true>
        <<<dim3(FF_ / 64, M_ / 128), 256, 0, stream>>>(
            h2_bf, wfc_bf, b_fc, nullptr, hf_bf, M_, FF_, D_);

    // ---- 8. Proj GEMM + residual(x1) -> d_out f32 ----
    gemm_bf16_kern<false, true, false>
        <<<dim3(D_ / 64, M_ / 128), 256, 0, stream>>>(
            hf_bf, wpr_bf, b_proj, x1_f, out, M_, D_, FF_);
}
